// PPOagent_57913339019477
// MI455X (gfx1250) — matrix operand order repaired and checked by reading, not compiled, not run on hardware
//
#include <hip/hip_runtime.h>
#include <hip/hip_bf16.h>

#define T_DIM 4096
#define N_ENV 256
#define D_IN  64
#define A_DIM 9
#define B_TOT (T_DIM * N_ENV)
#define GAMMA 0.99f
#define LAM   0.95f
#define CLIP  0.2f
#define ENT_COEF 0.01f
#define VF_COEF  0.5f

#define MAIN_BLOCKS 16384   // 4 waves/block * 16 rows/wave = 64 rows/block
#define STAT_BLOCKS 1024

typedef __attribute__((ext_vector_type(16))) _Float16 v16h;
typedef __attribute__((ext_vector_type(8)))  float    v8f;

// ---- fast transcendental helpers (CDNA5 TRANS ops, no libm branches) -------
__device__ __forceinline__ float fast_tanh(float x) {
#if __has_builtin(__builtin_amdgcn_tanhf)
  return __builtin_amdgcn_tanhf(x);            // v_tanh_f32
#else
  // tanh(x) = 1 - 2/(exp2(2*log2e*x)+1), branchless via v_exp_f32 + v_rcp_f32
  const float t = __builtin_amdgcn_exp2f(x * 2.8853900817779268f);
  return 1.0f - 2.0f * __builtin_amdgcn_rcpf(t + 1.0f);
#endif
}

__device__ __forceinline__ float fast_exp(float x) {
#if __has_builtin(__builtin_amdgcn_exp2f)
  return __builtin_amdgcn_exp2f(x * 1.4426950408889634f);   // v_exp_f32
#else
  return expf(x);
#endif
}

__device__ __forceinline__ float fast_log(float x) {
#if __has_builtin(__builtin_amdgcn_logf)
  return __builtin_amdgcn_logf(x) * 0.6931471805599453f;    // v_log_f32
#else
  return logf(x);
#endif
}

__device__ __forceinline__ v8f wmma16x16x32(v16h a, v16h b, v8f c) {
  return __builtin_amdgcn_wmma_f32_16x16x32_f16(false, a, false, b, (short)0, c,
                                                false, false);
}

// Wave-local ordering of LDS stores vs later LDS loads (single wave owns its
// tile buffers; DS ops are in-order per wave, the wait guarantees completion).
__device__ __forceinline__ void wave_sync_lds() {
  __builtin_amdgcn_wave_barrier();
  asm volatile("s_wait_dscnt 0" ::: "memory");
  __builtin_amdgcn_wave_barrier();
}

// ---- WMMA operand layouts (CDNA5 ISA 7.12.2, wave32) -----------------------
// A (16x32 f16): lane L holds row M=L&15; slot i -> K = (i&7) + ((i>>3)<<4) + ((L>>4)<<3)
// B (32x16 f16): lane L holds col N=L&15; slot i -> K = (i&7) + ((i>>3)<<4) + ((L>>4)<<3)
// C/D (16x16 f32): element r -> M = r + ((L>>4)<<3), N = L&15

// Convert f32 weights (row-major [64 x realCols]) into f16 fragment-swizzled
// LDS layout: frag = kt*nTiles + nt ; [frag][lane][slot] contiguous so a wave
// fetches one B fragment as a 32B (2x b128) LDS load.
__device__ __forceinline__ void fill_swz(const float* __restrict__ W, int realCols,
                                         _Float16* dst, int nTiles,
                                         int tid, int nthreads) {
  const int total = nTiles * 2 * 32 * 16;
  for (int idx = tid; idx < total; idx += nthreads) {
    const int slot = idx & 15;
    const int lane = (idx >> 4) & 31;
    const int frag = idx >> 9;
    const int kt = frag / nTiles;
    const int nt = frag - kt * nTiles;
    const int col = nt * 16 + (lane & 15);
    const int k = kt * 32 + (slot & 7) + ((slot >> 3) << 4) + ((lane >> 4) << 3);   // B-matrix K layout
    const float v = (col < realCols) ? W[k * realCols + col] : 0.0f;
    dst[idx] = (_Float16)v;
  }
}

__device__ __forceinline__ v16h ldfrag(const _Float16* s, int nTiles, int nt,
                                       int kt, int lane) {
  return *(const v16h*)(s + (((kt * nTiles + nt) * 32 + lane) << 4));
}

// A fragment straight from global f32 rows (two 8-float contiguous segments).
__device__ __forceinline__ v16h load_a_global(const float* __restrict__ rp,
                                              int lane, int kt) {
  const int base = kt * 32 + ((lane >> 4) << 3);
  const float4 f0 = *(const float4*)(rp + base);
  const float4 f1 = *(const float4*)(rp + base + 4);
  const float4 f2 = *(const float4*)(rp + base + 16);
  const float4 f3 = *(const float4*)(rp + base + 20);
  v16h a;
  a[0]=(_Float16)f0.x;  a[1]=(_Float16)f0.y;  a[2]=(_Float16)f0.z;  a[3]=(_Float16)f0.w;
  a[4]=(_Float16)f1.x;  a[5]=(_Float16)f1.y;  a[6]=(_Float16)f1.z;  a[7]=(_Float16)f1.w;
  a[8]=(_Float16)f2.x;  a[9]=(_Float16)f2.y;  a[10]=(_Float16)f2.z; a[11]=(_Float16)f2.w;
  a[12]=(_Float16)f3.x; a[13]=(_Float16)f3.y; a[14]=(_Float16)f3.z; a[15]=(_Float16)f3.w;
  return a;
}

// A fragment from row-major 16x64 f16 hidden tile in LDS.
__device__ __forceinline__ v16h load_a_lds(const _Float16* h, int lane, int kt) {
  const int row = lane & 15;
  const int base = kt * 32 + ((lane >> 4) << 3);
  const _Float16* p = h + row * 64 + base;
  v16h a;
#pragma unroll
  for (int i = 0; i < 8; ++i) { a[i] = p[i]; a[i + 8] = p[16 + i]; }
  return a;
}

// hidden = tanh(A[16x64] @ W[64x64] + b), written to row-major 16x64 f16 tile.
__device__ __forceinline__ void layer64(v16h a0, v16h a1, const _Float16* wsz,
                                        const float* bias, int lane,
                                        _Float16* hout) {
  const int M0 = (lane >> 4) << 3;
  const int nc = lane & 15;
#pragma unroll
  for (int nt = 0; nt < 4; ++nt) {
    const float bv = bias[nt * 16 + nc];
    v8f acc;
#pragma unroll
    for (int r = 0; r < 8; ++r) acc[r] = bv;
    acc = wmma16x16x32(a0, ldfrag(wsz, 4, nt, 0, lane), acc);
    acc = wmma16x16x32(a1, ldfrag(wsz, 4, nt, 1, lane), acc);
#pragma unroll
    for (int r = 0; r < 8; ++r)
      hout[(M0 + r) * 64 + nt * 16 + nc] = (_Float16)fast_tanh(acc[r]);
  }
}

// head = A[16x64] @ W[64x16pad] + b, written to 16x16 f32 tile (no tanh).
__device__ __forceinline__ void layer_head(v16h a0, v16h a1, const _Float16* wsz,
                                           const float* bias, int lane,
                                           float* outt) {
  const int M0 = (lane >> 4) << 3;
  const int nc = lane & 15;
  const float bv = bias[nc];
  v8f acc;
#pragma unroll
  for (int r = 0; r < 8; ++r) acc[r] = bv;
  acc = wmma16x16x32(a0, ldfrag(wsz, 1, 0, 0, lane), acc);
  acc = wmma16x16x32(a1, ldfrag(wsz, 1, 0, 1, lane), acc);
#pragma unroll
  for (int r = 0; r < 8; ++r) outt[(M0 + r) * 16 + nc] = acc[r];
}

// ---------------- Kernel 0: critic MLP on next_obs (256 rows, f32 VALU) -----
__global__ __launch_bounds__(256) void critic_next_kernel(
    const float* __restrict__ next_obs,
    const float* __restrict__ cW1, const float* __restrict__ cb1,
    const float* __restrict__ cW2, const float* __restrict__ cb2,
    const float* __restrict__ cW3, const float* __restrict__ cb3,
    float* __restrict__ nextval) {
  __shared__ float w1[64 * 64], w2[64 * 64], w3[64], b1[64], b2[64];
  const int tid = threadIdx.x;
  for (int i = tid; i < 4096; i += 256) { w1[i] = cW1[i]; w2[i] = cW2[i]; }
  for (int i = tid; i < 64; i += 256) { w3[i] = cW3[i]; b1[i] = cb1[i]; b2[i] = cb2[i]; }
  __syncthreads();
  float x[64], h[64];
  const float* rp = next_obs + tid * 64;
  for (int k = 0; k < 64; ++k) x[k] = rp[k];
  for (int j = 0; j < 64; ++j) {
    float s = b1[j];
    for (int k = 0; k < 64; ++k) s += x[k] * w1[k * 64 + j];
    h[j] = fast_tanh(s);
  }
  for (int j = 0; j < 64; ++j) {
    float s = b2[j];
    for (int k = 0; k < 64; ++k) s += h[k] * w2[k * 64 + j];
    x[j] = fast_tanh(s);
  }
  float s = cb3[0];
  for (int k = 0; k < 64; ++k) s += x[k] * w3[k];
  nextval[tid] = s;
}

// ---------------- Kernel 1: GAE reverse scan (256 independent columns) ------
__global__ __launch_bounds__(256) void gae_kernel(
    const float* __restrict__ rewards, const float* __restrict__ values,
    const int* __restrict__ dones, const int* __restrict__ next_done,
    const float* __restrict__ nextval,
    float* __restrict__ adv, float* __restrict__ ret) {
  const int n = threadIdx.x;
  float last = 0.0f;
  float v_next = nextval[n];
  float nnt_next = 1.0f - (float)next_done[n];
  for (int t = T_DIM - 1; t >= 0; --t) {
    const int i = t * N_ENV + n;
    const float vt = values[i];
    const float delta = rewards[i] + GAMMA * v_next * nnt_next - vt;
    const float a = delta + GAMMA * LAM * nnt_next * last;
    adv[i] = a;
    ret[i] = a + vt;
    last = a;
    v_next = vt;
    nnt_next = 1.0f - (float)dones[i];
  }
}

// ---------------- Kernels 2/3: advantage mean/std (ddof=1) ------------------
__global__ __launch_bounds__(256) void adv_stats_partial_kernel(
    const float* __restrict__ adv, float* __restrict__ part) {
  __shared__ float rs[256], rss[256];
  const int tid = threadIdx.x;
  float s = 0.0f, ss = 0.0f;
  for (int i = blockIdx.x * 256 + tid; i < B_TOT; i += 256 * STAT_BLOCKS) {
    const float v = adv[i];
    s += v; ss += v * v;
  }
  rs[tid] = s; rss[tid] = ss;
  __syncthreads();
  for (int off = 128; off > 0; off >>= 1) {
    if (tid < off) { rs[tid] += rs[tid + off]; rss[tid] += rss[tid + off]; }
    __syncthreads();
  }
  if (tid == 0) { part[blockIdx.x * 2] = rs[0]; part[blockIdx.x * 2 + 1] = rss[0]; }
}

__global__ __launch_bounds__(256) void adv_stats_finalize_kernel(
    const float* __restrict__ part, float* __restrict__ stats) {
  __shared__ float rs[256], rss[256];
  const int tid = threadIdx.x;
  float s = 0.0f, ss = 0.0f;
  for (int i = tid; i < STAT_BLOCKS; i += 256) { s += part[i * 2]; ss += part[i * 2 + 1]; }
  rs[tid] = s; rss[tid] = ss;
  __syncthreads();
  for (int off = 128; off > 0; off >>= 1) {
    if (tid < off) { rs[tid] += rs[tid + off]; rss[tid] += rss[tid + off]; }
    __syncthreads();
  }
  if (tid == 0) {
    const float mean = rs[0] / (float)B_TOT;
    const float var = (rss[0] - (float)B_TOT * mean * mean) / (float)(B_TOT - 1);
    stats[0] = mean;
    stats[1] = sqrtf(fmaxf(var, 0.0f));
  }
}

// ---------------- Kernel 4: fused actor+critic WMMA forward + PPO terms -----
__global__ __launch_bounds__(128) void ppo_fused_kernel(
    const float* __restrict__ obs, const float* __restrict__ logprobs,
    const float* __restrict__ values, const int* __restrict__ actions,
    const float* __restrict__ aW1, const float* __restrict__ ab1,
    const float* __restrict__ aW2, const float* __restrict__ ab2,
    const float* __restrict__ aW3, const float* __restrict__ ab3,
    const float* __restrict__ cW1, const float* __restrict__ cb1,
    const float* __restrict__ cW2, const float* __restrict__ cb2,
    const float* __restrict__ cW3, const float* __restrict__ cb3,
    const float* __restrict__ adv, const float* __restrict__ ret,
    const float* __restrict__ stats, float* __restrict__ partials) {
  __shared__ __align__(32) _Float16 s_aW1[4096];
  __shared__ __align__(32) _Float16 s_aW2[4096];
  __shared__ __align__(32) _Float16 s_aW3[1024];
  __shared__ __align__(32) _Float16 s_cW1[4096];
  __shared__ __align__(32) _Float16 s_cW2[4096];
  __shared__ __align__(32) _Float16 s_cW3[1024];
  __shared__ float s_b[288];
  __shared__ __align__(32) _Float16 s_h[4][16 * 64];
  __shared__ float s_out[4][256];
  __shared__ float s_acc[3];

  const int tid = threadIdx.x;
  fill_swz(aW1, 64, s_aW1, 4, tid, 128);
  fill_swz(aW2, 64, s_aW2, 4, tid, 128);
  fill_swz(aW3, A_DIM, s_aW3, 1, tid, 128);
  fill_swz(cW1, 64, s_cW1, 4, tid, 128);
  fill_swz(cW2, 64, s_cW2, 4, tid, 128);
  fill_swz(cW3, 1, s_cW3, 1, tid, 128);
  for (int i = tid; i < 288; i += 128) {
    float v = 0.0f;
    if (i < 64)       v = ab1[i];
    else if (i < 128) v = ab2[i - 64];
    else if (i < 144) v = (i - 128 < A_DIM) ? ab3[i - 128] : 0.0f;
    else if (i < 208) v = cb1[i - 144];
    else if (i < 272) v = cb2[i - 208];
    else              v = (i == 272) ? cb3[0] : 0.0f;
    s_b[i] = v;
  }
  if (tid < 3) s_acc[tid] = 0.0f;
  __syncthreads();

  const int wave = tid >> 5;
  const int lane = tid & 31;
  const int tile = blockIdx.x * 4 + wave;
  const int rowBase = tile * 16;
  _Float16* h = s_h[wave];
  float* outt = s_out[wave];
  const float* rp = obs + (size_t)(rowBase + (lane & 15)) * D_IN;

  // ------------------- actor -------------------
  v16h a0 = load_a_global(rp, lane, 0);
  v16h a1 = load_a_global(rp, lane, 1);
  layer64(a0, a1, s_aW1, s_b + 0, lane, h);
  wave_sync_lds();
  a0 = load_a_lds(h, lane, 0);
  a1 = load_a_lds(h, lane, 1);
  wave_sync_lds();
  layer64(a0, a1, s_aW2, s_b + 64, lane, h);
  wave_sync_lds();
  a0 = load_a_lds(h, lane, 0);
  a1 = load_a_lds(h, lane, 1);
  layer_head(a0, a1, s_aW3, s_b + 128, lane, outt);
  wave_sync_lds();

  float nlp = 0.0f, ent = 0.0f;
  if (lane < 16) {
    const int idx = rowBase + lane;
    const int act = actions[idx];
    float lg[A_DIM];
    float m = -1e30f;
#pragma unroll
    for (int j = 0; j < A_DIM; ++j) {
      lg[j] = outt[lane * 16 + j];
      m = fmaxf(m, lg[j]);
    }
    float se = 0.0f;
#pragma unroll
    for (int j = 0; j < A_DIM; ++j) se += fast_exp(lg[j] - m);
    const float lse = m + fast_log(se);
    float la = 0.0f, sp = 0.0f;
#pragma unroll
    for (int j = 0; j < A_DIM; ++j) {
      sp += fast_exp(lg[j] - lse) * lg[j];
      la = (j == act) ? lg[j] : la;
    }
    nlp = la - lse;
    ent = lse - sp;   // -sum p*logp
  }
  wave_sync_lds();

  // ------------------- critic -------------------
  a0 = load_a_global(rp, lane, 0);
  a1 = load_a_global(rp, lane, 1);
  layer64(a0, a1, s_cW1, s_b + 144, lane, h);
  wave_sync_lds();
  a0 = load_a_lds(h, lane, 0);
  a1 = load_a_lds(h, lane, 1);
  wave_sync_lds();
  layer64(a0, a1, s_cW2, s_b + 208, lane, h);
  wave_sync_lds();
  a0 = load_a_lds(h, lane, 0);
  a1 = load_a_lds(h, lane, 1);
  layer_head(a0, a1, s_cW3, s_b + 272, lane, outt);
  wave_sync_lds();

  if (lane < 16) {
    const int idx = rowBase + lane;
    const float nv = outt[lane * 16];             // critic column 0
    const float advv = adv[idx];
    const float retv = ret[idx];
    const float valv = values[idx];
    const float ratio = fast_exp(nlp - logprobs[idx]);
    const float mb = (advv - stats[0]) / (stats[1] + 1e-8f);
    const float rc = fminf(fmaxf(ratio, 1.0f - CLIP), 1.0f + CLIP);
    const float pg = fmaxf(-mb * ratio, -mb * rc);
    float vun = nv - retv; vun *= vun;
    const float vd = fminf(fmaxf(nv - valv, -CLIP), CLIP);
    float vcl = valv + vd - retv; vcl *= vcl;
    const float vm = fmaxf(vun, vcl);
    atomicAdd(&s_acc[0], pg);
    atomicAdd(&s_acc[1], ent);
    atomicAdd(&s_acc[2], vm);
  }
  __syncthreads();
  if (tid < 3) partials[blockIdx.x * 3 + tid] = s_acc[tid];
}

// ---------------- Kernel 5: final loss ---------------------------------------
__global__ __launch_bounds__(256) void finalize_loss_kernel(
    const float* __restrict__ partials, float* __restrict__ out) {
  __shared__ float r0[256], r1[256], r2[256];
  const int tid = threadIdx.x;
  float s0 = 0.0f, s1 = 0.0f, s2 = 0.0f;
  for (int i = tid; i < MAIN_BLOCKS; i += 256) {
    s0 += partials[i * 3];
    s1 += partials[i * 3 + 1];
    s2 += partials[i * 3 + 2];
  }
  r0[tid] = s0; r1[tid] = s1; r2[tid] = s2;
  __syncthreads();
  for (int off = 128; off > 0; off >>= 1) {
    if (tid < off) {
      r0[tid] += r0[tid + off];
      r1[tid] += r1[tid + off];
      r2[tid] += r2[tid + off];
    }
    __syncthreads();
  }
  if (tid == 0) {
    const float invB = 1.0f / (float)B_TOT;
    out[0] = r0[0] * invB - ENT_COEF * r1[0] * invB + VF_COEF * 0.5f * r2[0] * invB;
  }
}

extern "C" void kernel_launch(void* const* d_in, const int* in_sizes, int n_in,
                              void* d_out, int out_size, void* d_ws, size_t ws_size,
                              hipStream_t stream) {
  (void)in_sizes; (void)n_in; (void)out_size; (void)ws_size;
  const float* obs      = (const float*)d_in[0];
  const float* rewards  = (const float*)d_in[1];
  const float* values   = (const float*)d_in[2];
  const float* logprobs = (const float*)d_in[3];
  const float* next_obs = (const float*)d_in[4];
  const int* dones      = (const int*)d_in[5];
  const int* next_done  = (const int*)d_in[6];
  const int* actions    = (const int*)d_in[7];
  const float* aW1 = (const float*)d_in[8];
  const float* ab1 = (const float*)d_in[9];
  const float* aW2 = (const float*)d_in[10];
  const float* ab2 = (const float*)d_in[11];
  const float* aW3 = (const float*)d_in[12];
  const float* ab3 = (const float*)d_in[13];
  const float* cW1 = (const float*)d_in[14];
  const float* cb1 = (const float*)d_in[15];
  const float* cW2 = (const float*)d_in[16];
  const float* cb2 = (const float*)d_in[17];
  const float* cW3 = (const float*)d_in[18];
  const float* cb3 = (const float*)d_in[19];

  float* ws = (float*)d_ws;
  float* ws_nextval = ws;                       // 256
  float* ws_adv     = ws + 256;                 // B_TOT
  float* ws_ret     = ws_adv + B_TOT;           // B_TOT
  float* ws_statp   = ws_ret + B_TOT;           // STAT_BLOCKS*2
  float* ws_stats   = ws_statp + STAT_BLOCKS * 2; // 2 (mean, std)
  float* ws_part    = ws_stats + 2;             // MAIN_BLOCKS*3

  critic_next_kernel<<<1, 256, 0, stream>>>(next_obs, cW1, cb1, cW2, cb2, cW3, cb3,
                                            ws_nextval);
  gae_kernel<<<1, 256, 0, stream>>>(rewards, values, dones, next_done, ws_nextval,
                                    ws_adv, ws_ret);
  adv_stats_partial_kernel<<<STAT_BLOCKS, 256, 0, stream>>>(ws_adv, ws_statp);
  adv_stats_finalize_kernel<<<1, 256, 0, stream>>>(ws_statp, ws_stats);
  ppo_fused_kernel<<<MAIN_BLOCKS, 128, 0, stream>>>(
      obs, logprobs, values, actions,
      aW1, ab1, aW2, ab2, aW3, ab3,
      cW1, cb1, cW2, cb2, cW3, cb3,
      ws_adv, ws_ret, ws_stats, ws_part);
  finalize_loss_kernel<<<1, 256, 0, stream>>>(ws_part, (float*)d_out);
}
